// MultiHeadAttention2_13726715478445
// MI455X (gfx1250) — compile-verified, matrix operand order checked by reading, not run on hardware
//
#include <hip/hip_runtime.h>
#include <hip/hip_bf16.h>

typedef __bf16 bf16;
typedef __attribute__((ext_vector_type(16))) __bf16 v16bf;
typedef __attribute__((ext_vector_type(8)))  float  v8f;
typedef int v4i_ __attribute__((vector_size(16)));

#define D_MODEL 1024
#define SEQ     4096
#define NHEAD   16
#define DK      64

#define AS1 __attribute__((address_space(1)))
#define AS3 __attribute__((address_space(3)))

// ---------------------------------------------------------------------------
// CDNA5 async global->LDS copy (ASYNCcnt-tracked), with compile-safe fallback.
// Builtin signature (from toolchain diagnostic): (v4i AS1*, v4i AS3*, Ii, Ii).
// AS1 value == flat address; LDS pointer passed as its low 32 bits.
// ---------------------------------------------------------------------------
#if __has_builtin(__builtin_amdgcn_global_load_async_to_lds_b128)
#define ASYNC_CP 1
static __device__ __forceinline__ void async_cp16(const void* g, void* l) {
  __builtin_amdgcn_global_load_async_to_lds_b128(
      (AS1 v4i_*)(unsigned long long)(size_t)g,
      (AS3 v4i_*)(unsigned)(size_t)l, 0, 0);
}
static __device__ __forceinline__ void async_wait0() {
#if __has_builtin(__builtin_amdgcn_s_wait_asynccnt)
  __builtin_amdgcn_s_wait_asynccnt(0);
#else
  asm volatile("s_wait_asynccnt 0" ::: "memory");
#endif
}
#else
#define ASYNC_CP 0
#endif

// ---------------------------------------------------------------------------
// Fragment loader: 16x32 bf16 tile (A-layout, or B-layout from an [N][K]
// staged tile) from LDS, per CDNA5 ISA 7.12.2:
//   lanes 0-15 : M = lane,    K = {0..7, 16..23}
//   lanes 16-31: M = lane-16, K = {8..15, 24..31}
// Two 16-byte ds_load_b128 per lane.
// ---------------------------------------------------------------------------
static __device__ __forceinline__ v16bf ldfrag(const bf16* base, int ld) {
  const int lane = threadIdx.x & 31;
  const int m    = lane & 15;
  const int b0   = (lane & 16) ? 8 : 0;
  const bf16* p  = base + m * ld + b0;
  union { uint4 u[2]; v16bf v; } r;
  r.u[0] = *reinterpret_cast<const uint4*>(p);
  r.u[1] = *reinterpret_cast<const uint4*>(p + 16);
  return r.v;
}

static __device__ __forceinline__ v8f wmma_bf16(v16bf a, v16bf b, v8f c) {
  return __builtin_amdgcn_wmma_f32_16x16x32_bf16(false, a, false, b,
                                                 (short)0, c, false, false);
}

// ---------------------------------------------------------------------------
// Transpose + convert weights: Wt[i][o] = (bf16) W[o][i]
// ---------------------------------------------------------------------------
__global__ void k_prep_w(const float* __restrict__ W, bf16* __restrict__ Wt) {
  int t = blockIdx.x * 256 + threadIdx.x;      // 1M elements
  int o = t >> 10;
  int i = t & 1023;
  Wt[i * D_MODEL + o] = (bf16)W[t];
}

// ---------------------------------------------------------------------------
// Tiled GEMM: Y = X @ W^T + b
// MODE 0: A f32, bias+RoPE, *oscale, head-major bf16 [H][S][DK]  (Q with 1/8, K)
// MODE 1: A f32, bias,      *oscale, head-major bf16            (V)
// MODE 2: A bf16 (attn out), bias, f32 [S][D_MODEL]             (O-proj)
// Workgroup = 256 threads = 8 waves; tile 128 rows x 64 cols; K step 32.
// ---------------------------------------------------------------------------
template <int MODE>
__global__ void k_gemm(const float* __restrict__ Af,
                       const bf16*  __restrict__ Ab,
                       const bf16*  __restrict__ Wt,
                       const float* __restrict__ bias,
                       bf16*        __restrict__ outH,
                       float*       __restrict__ outF,
                       float        oscale) {
  __shared__ alignas(16) bf16 As[128 * 32];   // [row][k]
  __shared__ alignas(16) bf16 Bs[64 * 32];    // [n][k]

  const int tid  = threadIdx.x;
  const int lane = tid & 31;
  const int wave = tid >> 5;
  const int row0 = blockIdx.x * 128;
  const int col0 = blockIdx.y * 64;
  const int m0   = wave * 16;

  v8f c[4] = {};

  for (int kk = 0; kk < D_MODEL; kk += 32) {
#if ASYNC_CP
    if (MODE == 2) {
      // bf16 source, layout-preserving: async 16B chunks (2 per thread)
#pragma unroll
      for (int rep = 0; rep < 2; ++rep) {
        int e = rep * 2048 + tid * 8;          // element index into 128x32 tile
        int r = e >> 5, cc = e & 31;
        async_cp16(Ab + (size_t)(row0 + r) * D_MODEL + kk + cc, &As[e]);
      }
    }
#endif
    if (MODE != 2 || !ASYNC_CP) {
#pragma unroll
      for (int rep = 0; rep < 16; ++rep) {     // 128x32 A tile
        int idx = rep * 256 + tid;
        int r = idx >> 5, cc = idx & 31;
        if (MODE == 2) As[idx] = Ab[(size_t)(row0 + r) * D_MODEL + kk + cc];
        else           As[idx] = (bf16)Af[(size_t)(row0 + r) * D_MODEL + kk + cc];
      }
    }
#pragma unroll
    for (int rep = 0; rep < 8; ++rep) {        // 32x64 B tile -> [n][k]
      int idx = rep * 256 + tid;
      int k = idx >> 6, n = idx & 63;
      Bs[n * 32 + k] = Wt[(size_t)(kk + k) * D_MODEL + col0 + n];
    }
#if ASYNC_CP
    if (MODE == 2) async_wait0();
#endif
    __syncthreads();

    v16bf a = ldfrag(&As[m0 * 32], 32);
#pragma unroll
    for (int nt = 0; nt < 4; ++nt) {
      v16bf b = ldfrag(&Bs[(nt * 16) * 32], 32);
      c[nt] = wmma_bf16(a, b, c[nt]);
    }
    __syncthreads();
  }

  // Epilogue. C layout: lane holds column n = lane&15, rows
  // (lane<16 ? 0..7 : 8..15) in the 8 f32 elements.
  const int n  = lane & 15;
  const int rb = (lane >> 4) << 3;
#pragma unroll
  for (int nt = 0; nt < 4; ++nt) {
    const int col = col0 + nt * 16 + n;
    const float bv = bias[col];
#pragma unroll
    for (int v = 0; v < 8; ++v) c[nt][v] += bv;

    if (MODE == 2) {
#pragma unroll
      for (int v = 0; v < 8; ++v)
        outF[(size_t)(row0 + m0 + rb + v) * D_MODEL + col] = c[nt][v];
    } else {
      const int h = col >> 6;
      const int d = col & 63;
      if (MODE == 0) {
        // RoPE: pairs (2i, 2i+1) are adjacent columns -> adjacent lanes.
        const float freq = __powf(10000.0f, -(float)(d & ~1) * (1.0f / 64.0f));
        const bool  ev   = (d & 1) == 0;
#pragma unroll
        for (int v = 0; v < 8; ++v) {
          const int pos = row0 + m0 + rb + v;
          float sn, cs;
          __sincosf((float)pos * freq, &sn, &cs);
          float x  = c[nt][v];
          float xp = __shfl_xor(x, 1, 32);
          c[nt][v] = ev ? (x * cs - xp * sn) : (xp * sn + x * cs);
        }
      }
#pragma unroll
      for (int v = 0; v < 8; ++v) {
        const int pos = row0 + m0 + rb + v;
        outH[((size_t)h * SEQ + pos) * DK + d] = (bf16)(c[nt][v] * oscale);
      }
    }
  }
}

// ---------------------------------------------------------------------------
// Flash attention: one (head, 128-query-row) tile per workgroup, kv tiles of 32.
// 1/sqrt(dk) pre-folded into Q. Row-sum of P computed by an extra WMMA against
// an all-ones B fragment (lands directly in the l_i layout -> no shuffles);
// only the row-max still uses a 4-step shuffle tree.
// ---------------------------------------------------------------------------
__global__ void k_attn(const bf16* __restrict__ qh, const bf16* __restrict__ kh,
                       const bf16* __restrict__ vh, bf16* __restrict__ outA) {
  __shared__ alignas(16) bf16 Qs[128 * 64];   // [row][d]       16 KB
  __shared__ alignas(16) bf16 Ks[32 * 64];    // [kv][d]         4 KB
  __shared__ alignas(16) bf16 Vt[64 * 32];    // [d][kv]         4 KB
  __shared__ alignas(16) bf16 Ps[8][16 * 32]; // per-wave P      8 KB

  const int tid  = threadIdx.x;
  const int lane = tid & 31;
  const int wave = tid >> 5;
  const int row0 = blockIdx.x * 128;
  const int h    = blockIdx.y;
  const int m0   = wave * 16;

  const bf16* qb = qh + (size_t)h * SEQ * DK;
  const bf16* kb = kh + (size_t)h * SEQ * DK;
  const bf16* vb = vh + (size_t)h * SEQ * DK;

#if ASYNC_CP
#pragma unroll
  for (int rep = 0; rep < 4; ++rep) {          // 128x64 Q tile, 16B/thread/rep
    int e = rep * 2048 + tid * 8;
    async_cp16(qb + (size_t)row0 * DK + e, &Qs[e]);
  }
  async_wait0();
#else
#pragma unroll
  for (int rep = 0; rep < 32; ++rep) {
    int idx = rep * 256 + tid;
    Qs[idx] = qb[(size_t)row0 * DK + idx];
  }
#endif
  __syncthreads();
  const v16bf qa0 = ldfrag(&Qs[m0 * 64], 64);
  const v16bf qa1 = ldfrag(&Qs[m0 * 64 + 32], 64);

  v16bf ones;
#pragma unroll
  for (int i = 0; i < 16; ++i) ones[i] = (bf16)1.0f;

  v8f acc[4] = {};
  float m_i[8], l_i[8];
#pragma unroll
  for (int v = 0; v < 8; ++v) { m_i[v] = -3.0e38f; l_i[v] = 0.0f; }

  const int n  = lane & 15;
  const int rb = (lane >> 4) << 3;

  for (int j0 = 0; j0 < SEQ; j0 += 32) {
#if ASYNC_CP
    async_cp16(kb + (size_t)j0 * DK + tid * 8, &Ks[tid * 8]);  // 32x64 K tile
#endif
#pragma unroll
    for (int rep = 0; rep < 8; ++rep) {        // V^T staging (2B transpose)
      int idx = rep * 256 + tid;
      int r = idx >> 6, d = idx & 63;
#if !ASYNC_CP
      Ks[idx] = kb[(size_t)j0 * DK + idx];
#endif
      Vt[d * 32 + r] = vb[(size_t)j0 * DK + idx];
    }
    if (j0 + 32 < SEQ) {                       // global_prefetch_b8 next tiles
      __builtin_prefetch(kb + (size_t)(j0 + 32) * DK + tid * 8, 0, 0);
      __builtin_prefetch(vb + (size_t)(j0 + 32) * DK + tid * 8, 0, 0);
    }
#if ASYNC_CP
    async_wait0();
#endif
    __syncthreads();

    // S(16x32) = Q(16x64) * K^T  (scale already folded into Q)
    v8f s0 = {}, s1 = {};
    s0 = wmma_bf16(qa0, ldfrag(&Ks[0], 64), s0);
    s0 = wmma_bf16(qa1, ldfrag(&Ks[32], 64), s0);
    s1 = wmma_bf16(qa0, ldfrag(&Ks[16 * 64], 64), s1);
    s1 = wmma_bf16(qa1, ldfrag(&Ks[16 * 64 + 32], 64), s1);

    // row max across the 16 lanes sharing each C-tile row; rescale factor
    float alpha[8];
#pragma unroll
    for (int v = 0; v < 8; ++v) {
      float m = fmaxf(s0[v], s1[v]);
#pragma unroll
      for (int msk = 1; msk < 16; msk <<= 1)
        m = fmaxf(m, __shfl_xor(m, msk, 32));
      float mn = fmaxf(m_i[v], m);
      alpha[v] = __expf(m_i[v] - mn);
      m_i[v]   = mn;
    }
#pragma unroll
    for (int v = 0; v < 8; ++v) {
      s0[v] = __expf(s0[v] - m_i[v]);
      s1[v] = __expf(s1[v] - m_i[v]);
    }
#pragma unroll
    for (int nt = 0; nt < 4; ++nt)
#pragma unroll
      for (int v = 0; v < 8; ++v) acc[nt][v] *= alpha[v];

    // C-layout -> A-layout for P via per-wave LDS scratch (in-order DS ops)
#pragma unroll
    for (int v = 0; v < 8; ++v) {
      Ps[wave][(rb + v) * 32 + n]      = (bf16)s0[v];
      Ps[wave][(rb + v) * 32 + n + 16] = (bf16)s1[v];
    }
    v16bf pa = ldfrag(&Ps[wave][0], 32);

    // row-sum(P) via WMMA against all-ones B: lands in l_i's layout
    v8f zero = {};
    v8f ls = wmma_bf16(pa, ones, zero);
#pragma unroll
    for (int v = 0; v < 8; ++v) l_i[v] = l_i[v] * alpha[v] + ls[v];

#pragma unroll
    for (int nt = 0; nt < 4; ++nt)
      acc[nt] = wmma_bf16(pa, ldfrag(&Vt[(nt * 16) * 32], 32), acc[nt]);
    __syncthreads();
  }

  // normalize and store merged-head bf16 [S][D_MODEL]
#pragma unroll
  for (int v = 0; v < 8; ++v) {
    const float inv = 1.0f / l_i[v];
    const int   pos = row0 + m0 + rb + v;
#pragma unroll
    for (int nt = 0; nt < 4; ++nt)
      outA[(size_t)pos * D_MODEL + h * DK + nt * 16 + n] =
          (bf16)(acc[nt][v] * inv);
  }
}

// ---------------------------------------------------------------------------
extern "C" void kernel_launch(void* const* d_in, const int* in_sizes, int n_in,
                              void* d_out, int out_size, void* d_ws,
                              size_t ws_size, hipStream_t stream) {
  const float* Q  = (const float*)d_in[0];
  const float* K  = (const float*)d_in[1];
  const float* V  = (const float*)d_in[2];
  const float* Wq = (const float*)d_in[3];
  const float* bq = (const float*)d_in[4];
  const float* Wk = (const float*)d_in[5];
  const float* bk = (const float*)d_in[6];
  const float* Wv = (const float*)d_in[7];
  const float* bv = (const float*)d_in[8];
  const float* Wo = (const float*)d_in[9];
  const float* bo = (const float*)d_in[10];
  float* out = (float*)d_out;

  char* ws = (char*)d_ws;
  const size_t WB = (size_t)D_MODEL * D_MODEL * sizeof(bf16); // 2 MB
  const size_t TB = (size_t)SEQ * D_MODEL * sizeof(bf16);     // 8 MB
  bf16* wq_t = (bf16*)(ws);
  bf16* wk_t = (bf16*)(ws + WB);
  bf16* wv_t = (bf16*)(ws + 2 * WB);
  bf16* wo_t = (bf16*)(ws + 3 * WB);
  bf16* qh   = (bf16*)(ws + 4 * WB);
  bf16* kh   = (bf16*)(ws + 4 * WB + TB);
  bf16* vh   = (bf16*)(ws + 4 * WB + 2 * TB);
  bf16* at   = (bf16*)(ws + 4 * WB + 3 * TB);

  dim3 blk(256);
  k_prep_w<<<dim3((D_MODEL * D_MODEL) / 256), blk, 0, stream>>>(Wq, wq_t);
  k_prep_w<<<dim3((D_MODEL * D_MODEL) / 256), blk, 0, stream>>>(Wk, wk_t);
  k_prep_w<<<dim3((D_MODEL * D_MODEL) / 256), blk, 0, stream>>>(Wv, wv_t);
  k_prep_w<<<dim3((D_MODEL * D_MODEL) / 256), blk, 0, stream>>>(Wo, wo_t);

  dim3 grd(SEQ / 128, D_MODEL / 64);          // 32 x 16
  k_gemm<0><<<grd, blk, 0, stream>>>(Q, nullptr, wq_t, bq, qh, nullptr, 0.125f);
  k_gemm<0><<<grd, blk, 0, stream>>>(K, nullptr, wk_t, bk, kh, nullptr, 1.0f);
  k_gemm<1><<<grd, blk, 0, stream>>>(V, nullptr, wv_t, bv, vh, nullptr, 1.0f);

  k_attn<<<dim3(SEQ / 128, NHEAD), blk, 0, stream>>>(qh, kh, vh, at);

  k_gemm<2><<<grd, blk, 0, stream>>>(nullptr, at, wo_t, bo, nullptr, out, 1.0f);
}
